// FlowGCN_44143673868909
// MI455X (gfx1250) — compile-verified, matrix operand order checked by reading, not run on hardware
//
#include <hip/hip_runtime.h>

typedef __attribute__((ext_vector_type(16))) _Float16 v16h;
typedef __attribute__((ext_vector_type(8)))  float    v8f;

#define IN_DIM 128
#define HID    32

// ---------------------------------------------------------------------------
// 0) init: deg=1 (self loop), agg1=0, out=b2
// ---------------------------------------------------------------------------
__global__ void gcn_init_kernel(float* __restrict__ deg, float* __restrict__ agg1,
                                float* __restrict__ out, const float* __restrict__ b2,
                                int n_nodes) {
    long long i = (long long)blockIdx.x * blockDim.x + threadIdx.x;
    long long tot = (long long)n_nodes * HID;
    if (i < tot)      agg1[i] = 0.0f;
    if (i < n_nodes) { deg[i] = 1.0f; out[i] = b2[0]; }
}

// ---------------------------------------------------------------------------
// 1) degree accumulation over dst (edge_index row 1)
// ---------------------------------------------------------------------------
__global__ void gcn_deg_kernel(const int* __restrict__ ei, float* __restrict__ deg,
                               int n_edges) {
    int e = blockIdx.x * blockDim.x + threadIdx.x;
    if (e >= n_edges) return;
    atomicAdd(&deg[ei[n_edges + e]], 1.0f);
}

// ---------------------------------------------------------------------------
// 2) dinv = deg^-1/2 in place (deg >= 1 always, self loops included)
// ---------------------------------------------------------------------------
__global__ void gcn_rsqrt_kernel(float* __restrict__ deg, int n_nodes) {
    int i = blockIdx.x * blockDim.x + threadIdx.x;
    if (i >= n_nodes) return;
    deg[i] = rsqrtf(deg[i]);
}

// ---------------------------------------------------------------------------
// 3a) pack W1 (128x32 f32) into the 8 distinct WMMA B fragments, f16,
//     laid out exactly as the wave32 B VGPR image: [frag][lane][16 halves].
//     frag = kb*2 + nt  (kb: K-block of 32, nt: 16-col tile).
//     One 256-thread block: thread = frag*32 + lane. 8 KB total -> L2 resident.
// ---------------------------------------------------------------------------
__global__ void gcn_pack_w1_kernel(const float* __restrict__ W1,
                                   _Float16* __restrict__ w1h) {
    int t    = threadIdx.x;           // 0..255
    int frag = t >> 5;                // 0..7
    int lane = t & 31;
    int kb = frag >> 1, nt = frag & 1;
    int r  = lane & 15;               // B column within tile
    int hi = lane >> 4;               // lane-half selects K sub-block
    int k0 = kb * 32 + hi * 8;
    _Float16* dst = w1h + ((long long)frag * 32 + lane) * 16;
    #pragma unroll
    for (int j = 0; j < 8; ++j) {
        dst[j]     = (_Float16)W1[(k0 + j)      * HID + nt * 16 + r];
        dst[8 + j] = (_Float16)W1[(k0 + 16 + j) * HID + nt * 16 + r];
    }
}

// ---------------------------------------------------------------------------
// 3b) h = x @ W1 via v_wmma_f32_16x16x32_f16, f32 accumulate.
//     One wave per 16-node tile (N = 6250*16 exactly). K=128 in 4 steps,
//     two 16-col B tiles cover HID=32. B fragments come prepacked from 3a
//     (two global_load_b128 per fragment); A loads are float4 (b128).
//     Wave32 VGPR layouts per CDNA5 ISA 7.12.2:
//       A: lane<16 -> row=lane,    halves 0..7=K0..7,  8..15=K16..23
//          lane>=16 -> row=lane-16, halves 0..7=K8..15, 8..15=K24..31
//       D: vgpr i, lane l -> row = i + 8*(l>>4), col = l&15
// ---------------------------------------------------------------------------
__global__ void gcn_gemm1_wmma_kernel(const float* __restrict__ x,
                                      const _Float16* __restrict__ w1h,
                                      float* __restrict__ h, int n_nodes) {
    int wave = (int)((blockIdx.x * (long long)blockDim.x + threadIdx.x) >> 5);
    int lane = threadIdx.x & 31;
    int row0 = wave * 16;
    if (row0 >= n_nodes) return;            // wave-uniform: EXEC stays all-1s

    int r  = lane & 15;
    int hi = lane >> 4;

    v8f acc0 = {};
    v8f acc1 = {};

    const float* xrow = x + (long long)(row0 + r) * IN_DIM;
    const v16h*  Bf   = (const v16h*)w1h;

    #pragma unroll
    for (int kb = 0; kb < 4; ++kb) {
        int k0 = kb * 32 + hi * 8;          // first K octet for this lane-half

        float4 q0 = *(const float4*)(xrow + k0);
        float4 q1 = *(const float4*)(xrow + k0 + 4);
        float4 q2 = *(const float4*)(xrow + k0 + 16);
        float4 q3 = *(const float4*)(xrow + k0 + 20);

        v16h a;
        a[0] = (_Float16)q0.x;  a[1] = (_Float16)q0.y;
        a[2] = (_Float16)q0.z;  a[3] = (_Float16)q0.w;
        a[4] = (_Float16)q1.x;  a[5] = (_Float16)q1.y;
        a[6] = (_Float16)q1.z;  a[7] = (_Float16)q1.w;
        a[8]  = (_Float16)q2.x; a[9]  = (_Float16)q2.y;
        a[10] = (_Float16)q2.z; a[11] = (_Float16)q2.w;
        a[12] = (_Float16)q3.x; a[13] = (_Float16)q3.y;
        a[14] = (_Float16)q3.z; a[15] = (_Float16)q3.w;

        v16h b0 = Bf[(kb * 2 + 0) * 32 + lane];   // prepacked fragment, 32 B
        v16h b1 = Bf[(kb * 2 + 1) * 32 + lane];

        acc0 = __builtin_amdgcn_wmma_f32_16x16x32_f16(
                   false, a, false, b0, (short)0, acc0, false, false);
        acc1 = __builtin_amdgcn_wmma_f32_16x16x32_f16(
                   false, a, false, b1, (short)0, acc1, false, false);
    }

    #pragma unroll
    for (int i = 0; i < 8; ++i) {
        int row = row0 + i + 8 * hi;
        h[(long long)row * HID + r]      = acc0[i];
        h[(long long)row * HID + 16 + r] = acc1[i];
    }
}

// ---------------------------------------------------------------------------
// 4) layer-1 aggregation: one wave per edge, one lane per feature.
//    agg1[dst][f] += h[src][f] * dinv[src]*dinv[dst]
//    Virtual edges e in [E, E+N) are the self loops.
//    agg1 (12.8 MB) and h (12.8 MB) are L2-resident -> atomics resolve in L2.
// ---------------------------------------------------------------------------
__global__ void gcn_agg1_kernel(const int* __restrict__ ei,
                                const float* __restrict__ h,
                                const float* __restrict__ dinv,
                                float* __restrict__ agg1,
                                int n_edges, int n_nodes) {
    long long tid = (long long)blockIdx.x * blockDim.x + threadIdx.x;
    long long tot = (long long)(n_edges + n_nodes) * HID;
    if (tid >= tot) return;
    int e = (int)(tid >> 5);
    int f = (int)(tid & 31);
    int s, d;
    if (e < n_edges) { s = ei[e]; d = ei[n_edges + e]; }
    else             { s = d = e - n_edges; }
    float nrm = dinv[s] * dinv[d];
    atomicAdd(&agg1[(long long)d * HID + f], h[(long long)s * HID + f] * nrm);
}

// ---------------------------------------------------------------------------
// 5) fused: h1 = relu(agg1 + b1); z = h1 @ W2  (dot-32 per node)
//    One wave per node, wave32 shfl_xor tree reduction.
// ---------------------------------------------------------------------------
__global__ void gcn_relu_dot_kernel(const float* __restrict__ agg1,
                                    const float* __restrict__ b1,
                                    const float* __restrict__ W2,
                                    float* __restrict__ z, int n_nodes) {
    long long tid = (long long)blockIdx.x * blockDim.x + threadIdx.x;
    int n = (int)(tid >> 5);
    int f = (int)(tid & 31);
    if (n >= n_nodes) return;
    float v = agg1[(long long)n * HID + f] + b1[f];
    v = v > 0.0f ? v : 0.0f;
    float p = v * W2[f];
    #pragma unroll
    for (int m = 16; m > 0; m >>= 1)
        p += __shfl_xor(p, m, 32);
    if (f == 0) z[n] = p;
}

// ---------------------------------------------------------------------------
// 6) layer-2 aggregation: out[dst] += z[src] * norm  (1 thread / edge)
//    out was pre-initialized to b2.
// ---------------------------------------------------------------------------
__global__ void gcn_agg2_kernel(const int* __restrict__ ei,
                                const float* __restrict__ z,
                                const float* __restrict__ dinv,
                                float* __restrict__ out,
                                int n_edges, int n_nodes) {
    int e = blockIdx.x * blockDim.x + threadIdx.x;
    if (e >= n_edges + n_nodes) return;
    int s, d;
    if (e < n_edges) { s = ei[e]; d = ei[n_edges + e]; }
    else             { s = d = e - n_edges; }
    atomicAdd(&out[d], z[s] * dinv[s] * dinv[d]);
}

// ---------------------------------------------------------------------------
extern "C" void kernel_launch(void* const* d_in, const int* in_sizes, int n_in,
                              void* d_out, int out_size, void* d_ws, size_t ws_size,
                              hipStream_t stream) {
    const float* x  = (const float*)d_in[0];
    const int*   ei = (const int*)  d_in[1];   // [2, E] flat: row0=src, row1=dst
    const float* W1 = (const float*)d_in[2];
    const float* b1 = (const float*)d_in[3];
    const float* W2 = (const float*)d_in[4];
    const float* b2 = (const float*)d_in[5];
    float*       out = (float*)d_out;

    const int N = in_sizes[0] / IN_DIM;        // 100000
    const int E = in_sizes[1] / 2;             // 3200000

    // workspace (floats): [0,N) deg/dinv | [N,33N) h | [33N,65N) agg1 |
    //                     [65N,66N) z    | [66N, 66N+2048) packed W1 f16 (8 KB)
    float*      ws   = (float*)d_ws;
    float*      deg  = ws;
    float*      h    = ws + (long long)N;
    float*      agg1 = ws + (long long)N * 33;
    float*      z    = ws + (long long)N * 65;
    _Float16*   w1h  = (_Float16*)(ws + (long long)N * 66);

    const int B = 256;

    long long initT = (long long)N * HID;
    gcn_init_kernel<<<(unsigned)((initT + B - 1) / B), B, 0, stream>>>(deg, agg1, out, b2, N);

    gcn_deg_kernel<<<(E + B - 1) / B, B, 0, stream>>>(ei, deg, E);

    gcn_rsqrt_kernel<<<(N + B - 1) / B, B, 0, stream>>>(deg, N);

    gcn_pack_w1_kernel<<<1, 256, 0, stream>>>(W1, w1h);

    long long gemmT = (long long)((N + 15) / 16) * 32;   // one wave per 16-row tile
    gcn_gemm1_wmma_kernel<<<(unsigned)((gemmT + B - 1) / B), B, 0, stream>>>(x, w1h, h, N);

    long long agg1T = (long long)(E + N) * HID;
    gcn_agg1_kernel<<<(unsigned)((agg1T + B - 1) / B), B, 0, stream>>>(ei, h, deg, agg1, E, N);

    long long rdT = (long long)N * 32;
    gcn_relu_dot_kernel<<<(unsigned)((rdT + B - 1) / B), B, 0, stream>>>(agg1, b1, W2, z, N);

    gcn_agg2_kernel<<<(E + N + B - 1) / B, B, 0, stream>>>(ei, z, deg, out, E, N);
}
